// GINEncoder_70806830841987
// MI455X (gfx1250) — matrix operand order repaired and checked by reading, not compiled, not run on hardware
//
#include <hip/hip_runtime.h>

typedef float    v8f  __attribute__((ext_vector_type(8)));
typedef _Float16 v16h __attribute__((ext_vector_type(16)));
typedef _Float16 v8h  __attribute__((ext_vector_type(8)));

#define HID 64

// ---------------- utility: zero a float buffer ----------------
__global__ void __launch_bounds__(256) zero_f32(float* __restrict__ p, int n) {
    int i = blockIdx.x * 256 + threadIdx.x;
    if (i < n) p[i] = 0.0f;
}

// ---------------- layer-0 scatter: 2-dim features ----------------
__global__ void __launch_bounds__(256) scatter_add2(
    const float* __restrict__ feat, const int* __restrict__ src,
    const int* __restrict__ dst, float* __restrict__ agg, int nedges) {
    int e = blockIdx.x * 256 + threadIdx.x;
    if (e < nedges) {
        int s = src[e], d = dst[e];
        atomicAdd(&agg[d * 2 + 0], feat[s * 2 + 0]);
        atomicAdd(&agg[d * 2 + 1], feat[s * 2 + 1]);
    }
}

// ---------------- 64-dim scatter: 16 threads per edge, float4 each ----------------
__global__ void __launch_bounds__(256) scatter_add64(
    const float* __restrict__ h, const int* __restrict__ src,
    const int* __restrict__ dst, float* __restrict__ agg, int nedges) {
    int tid = blockIdx.x * 256 + threadIdx.x;
    int e  = tid >> 4;
    int c4 = (tid & 15) * 4;
    if (e < nedges) {
        int s = src[e], d = dst[e];
        float4 v = *(const float4*)(h + (long long)s * HID + c4);
        float* a = agg + (long long)d * HID + c4;
        atomicAdd(a + 0, v.x);
        atomicAdd(a + 1, v.y);
        atomicAdd(a + 2, v.z);
        atomicAdd(a + 3, v.w);
    }
}

// ---------------- layer-0 first linear: K=2, scalar ----------------
__global__ void __launch_bounds__(256) lin0_relu(
    const float* __restrict__ feat, const float* __restrict__ agg0,
    const float* __restrict__ w1, const float* __restrict__ b1,
    float* __restrict__ out, int nrows) {
    int tid = blockIdx.x * 256 + threadIdx.x;
    int node = tid >> 6;
    int j = tid & 63;
    if (node < nrows) {
        float z0 = feat[node * 2 + 0] + agg0[node * 2 + 0];
        float z1 = feat[node * 2 + 1] + agg0[node * 2 + 1];
        float v = fmaf(z0, w1[j], fmaf(z1, w1[HID + j], b1[j]));
        out[(long long)node * HID + j] = v > 0.0f ? v : 0.0f;
    }
}

// ---------------- WMMA GEMM: Out = relu((A [+A2]) @ W + bias), K=N=64 ----------------
// Block = 128 threads = 4 wave32s; block covers 64 rows; each wave a 16-row slab.
__global__ void __launch_bounds__(128) gemm64_bias_relu(
    const float* __restrict__ A, const float* __restrict__ A2,
    const float* __restrict__ W,     // [64 x 64] row-major (k, n)
    const float* __restrict__ bias,  // [64]
    float* __restrict__ Out, int nrows) {
    __shared__ __align__(16) _Float16 ldsA[HID * HID];   // [row][k]  8 KB
    __shared__ __align__(16) _Float16 ldsWt[HID * HID];  // [n][k]    8 KB (transposed)

    const int t = threadIdx.x;
    const int row0 = blockIdx.x * 64;

    // Stage activations (fused optional add), fp32 -> f16
    for (int idx = t; idx < HID * HID; idx += 128) {
        int r = idx >> 6, c = idx & 63;
        int gr = row0 + r;
        float v = 0.0f;
        if (gr < nrows) {
            long long off = (long long)gr * HID + c;
            v = A[off];
            if (A2) v += A2[off];
        }
        ldsA[idx] = (_Float16)v;
    }
    // Stage weights transposed: ldsWt[n][k] = W[k][n]
    for (int idx = t; idx < HID * HID; idx += 128) {
        int k = idx >> 6, n = idx & 63;
        ldsWt[n * HID + k] = (_Float16)W[idx];
    }
    __syncthreads();

    const int wave = t >> 5;       // wave32
    const int lane = t & 31;
    const int nl = lane & 15;
    const int half = lane >> 4;
    const int m0 = wave * 16;      // row slab within block

    v8f acc[4] = {};               // 4 N-tiles of 16

    #pragma unroll
    for (int ks = 0; ks < 2; ++ks) {
        const int k0 = ks * 32;
        // A fragment: element i -> K = k0 + (i>>3)*16 + half*8 + (i&7), M = nl
        const _Float16* ap = ldsA + (m0 + nl) * HID + k0 + half * 8;
        v8h alo = *(const v8h*)(ap);
        v8h ahi = *(const v8h*)(ap + 16);
        v16h afrag = __builtin_shufflevector(alo, ahi,
            0, 1, 2, 3, 4, 5, 6, 7, 8, 9, 10, 11, 12, 13, 14, 15);
        #pragma unroll
        for (int nt = 0; nt < 4; ++nt) {
            // B fragment: element i -> K = k0 + (i>>3)*16 + half*8 + (i&7), N = nt*16 + nl
            const _Float16* wp = ldsWt + (nt * 16 + nl) * HID + k0 + half * 8;
            v8h blo = *(const v8h*)(wp);
            v8h bhi = *(const v8h*)(wp + 16);
            v16h bfrag = __builtin_shufflevector(blo, bhi,
                0, 1, 2, 3, 4, 5, 6, 7, 8, 9, 10, 11, 12, 13, 14, 15);
            acc[nt] = __builtin_amdgcn_wmma_f32_16x16x32_f16(
                false, afrag, false, bfrag, (short)0, acc[nt], false, false);
        }
    }

    // Epilogue: D VGPR r -> M = r + 8*half, N = nl (within the N-tile)
    #pragma unroll
    for (int nt = 0; nt < 4; ++nt) {
        const int col = nt * 16 + nl;
        const float bv = bias[col];
        #pragma unroll
        for (int r = 0; r < 8; ++r) {
            int gr = row0 + m0 + r + 8 * half;
            if (gr < nrows) {
                float v = acc[nt][r] + bv;
                Out[(long long)gr * HID + col] = v > 0.0f ? v : 0.0f;
            }
        }
    }
}

// ---------------- final column mean: one block per column ----------------
__global__ void __launch_bounds__(256) mean_cols(
    const float* __restrict__ h, float* __restrict__ out, int nrows) {
    __shared__ float red[256];
    const int c = blockIdx.x;  // 0..63
    float s = 0.0f;
    for (int r = threadIdx.x; r < nrows; r += 256)
        s += h[(long long)r * HID + c];
    red[threadIdx.x] = s;
    __syncthreads();
    for (int w = 128; w > 0; w >>= 1) {
        if (threadIdx.x < w) red[threadIdx.x] += red[threadIdx.x + w];
        __syncthreads();
    }
    if (threadIdx.x == 0) out[c] = red[0] / (float)nrows;
}

extern "C" void kernel_launch(void* const* d_in, const int* in_sizes, int n_in,
                              void* d_out, int out_size, void* d_ws, size_t ws_size,
                              hipStream_t stream) {
    const float* feat = (const float*)d_in[0];
    const int*   src  = (const int*)d_in[1];
    const int*   dst  = (const int*)d_in[2];
    const int n_nodes = in_sizes[0] / 2;   // IN_DIM = 2
    const int n_edges = in_sizes[1];

    const float *w1[3], *b1[3], *w2[3], *b2[3];
    for (int l = 0; l < 3; ++l) {
        w1[l] = (const float*)d_in[3 + 4 * l + 0];
        b1[l] = (const float*)d_in[3 + 4 * l + 1];
        w2[l] = (const float*)d_in[3 + 4 * l + 2];
        b2[l] = (const float*)d_in[3 + 4 * l + 3];
    }

    float* hbuf   = (float*)d_ws;                    // [n_nodes x 64]
    float* tbuf   = hbuf + (size_t)n_nodes * HID;    // [n_nodes x 64]
    float* aggbuf = tbuf + (size_t)n_nodes * HID;    // [n_nodes x 64]

    const int gemm_grid = (n_nodes + 63) / 64;

    // ---- layer 0 (d_in = 2) ----
    zero_f32<<<(n_nodes * 2 + 255) / 256, 256, 0, stream>>>(aggbuf, n_nodes * 2);
    scatter_add2<<<(n_edges + 255) / 256, 256, 0, stream>>>(feat, src, dst, aggbuf, n_edges);
    lin0_relu<<<(n_nodes * HID + 255) / 256, 256, 0, stream>>>(feat, aggbuf, w1[0], b1[0], tbuf, n_nodes);
    gemm64_bias_relu<<<gemm_grid, 128, 0, stream>>>(tbuf, nullptr, w2[0], b2[0], hbuf, n_nodes);

    // ---- layers 1, 2 (d_in = 64) ----
    for (int l = 1; l < 3; ++l) {
        zero_f32<<<(n_nodes * HID + 255) / 256, 256, 0, stream>>>(aggbuf, n_nodes * HID);
        scatter_add64<<<((n_edges * 16) + 255) / 256, 256, 0, stream>>>(hbuf, src, dst, aggbuf, n_edges);
        gemm64_bias_relu<<<gemm_grid, 128, 0, stream>>>(hbuf, aggbuf, w1[l], b1[l], tbuf, n_nodes);
        gemm64_bias_relu<<<gemm_grid, 128, 0, stream>>>(tbuf, nullptr, w2[l], b2[l], hbuf, n_nodes);
    }

    // ---- graph mean readout ----
    mean_cols<<<HID, 256, 0, stream>>>(hbuf, (float*)d_out, n_nodes);
}
